// GATModel_16518444221033
// MI455X (gfx1250) — matrix operand order repaired and checked by reading, not compiled, not run on hardware
//
#include <hip/hip_runtime.h>
#include <hip/hip_bf16.h>

// ---------------- problem constants ----------------
#define NN      50000
#define IN_DIM  128
#define HID     32
#define HEADS   4
#define F1      (HEADS * HID)   // 128
#define OUT_DIM 64
#define NE      800000
#define EE      (NE + NN)       // edges + self loops = 850000
#define NP      500000
#define BN_EPS  1e-5f
#define NEG_SLOPE 0.2f

typedef __attribute__((ext_vector_type(16))) _Float16 v16h;
typedef __attribute__((ext_vector_type(8)))  _Float16 v8h;
typedef __attribute__((ext_vector_type(8)))  float    v8f;

// ---------------- helpers ----------------
__device__ __forceinline__ unsigned encf(float f) {
  unsigned u = __float_as_uint(f);
  return (u >> 31) ? ~u : (u | 0x80000000u);
}
__device__ __forceinline__ float decf(unsigned u) {
  return __uint_as_float((u >> 31) ? (u & 0x7fffffffu) : ~u);
}

// A fragment (16x32 f16, MxK): lanes0-15 M=lane, halves0-7 = K[kb..kb+7] (+8 if lane>=16),
// halves8-15 = K[kb+16..] (+8 if lane>=16). Row-major A, ld = K.
__device__ __forceinline__ v16h load_a_frag(const _Float16* A, int ld, int row, int kb, int lane) {
  int kg = ((lane >> 4) << 3);
  const _Float16* p = A + (size_t)row * ld + kb + kg;
  v8h lo = *(const v8h*)p;
  v8h hi = *(const v8h*)(p + 16);
  return __builtin_shufflevector(lo, hi, 0,1,2,3,4,5,6,7,8,9,10,11,12,13,14,15);
}
// B fragment (32x16 f16, KxN) from transposed weights Bt[N][K]:
// halves0-7 = K[kb..kb+7] (+8 if lane>=16), halves8-15 = K[kb+16..] (+8 if lane>=16), matching A.
__device__ __forceinline__ v16h load_b_frag(const _Float16* Bt, int ld, int col, int kb, int lane) {
  const _Float16* p = Bt + (size_t)col * ld + kb + ((lane >> 4) << 3);
  v8h lo = *(const v8h*)p;
  v8h hi = *(const v8h*)(p + 16);
  return __builtin_shufflevector(lo, hi, 0,1,2,3,4,5,6,7,8,9,10,11,12,13,14,15);
}

// ---------------- conversion kernels ----------------
__global__ void cvt_f16(const float* __restrict__ s, _Float16* __restrict__ d, int n) {
  int i = blockIdx.x * blockDim.x + threadIdx.x;
  if (i < n) d[i] = (_Float16)s[i];
}
// Wt[n*K+k] = (f16) W[k*N+n]
__global__ void wtrans_f16(const float* __restrict__ W, _Float16* __restrict__ Wt, int K, int Nc) {
  int i = blockIdx.x * blockDim.x + threadIdx.x;
  if (i >= K * Nc) return;
  int n = i / K, k = i % K;
  Wt[i] = (_Float16)W[(size_t)k * Nc + n];
}

// ---------------- WMMA GEMM: C[M x Nc] = A_f16[M x K] * Bt_f16[Nc x K]^T ----------------
__global__ void __launch_bounds__(128) wmma_gemm(const _Float16* __restrict__ A,
                                                 const _Float16* __restrict__ Bt,
                                                 float* __restrict__ C,
                                                 int M, int Nc, int K) {
  int wave = blockIdx.x * (blockDim.x >> 5) + (threadIdx.x >> 5);
  int lane = threadIdx.x & 31;
  int ntn = Nc >> 4;
  if (wave >= (M >> 4) * ntn) return;
  int mt = wave / ntn, nt = wave % ntn;
  int arow = (mt << 4) + (lane & 15);
  int bcol = (nt << 4) + (lane & 15);
  v8f c = {};
  for (int kb = 0; kb < K; kb += 32) {
    v16h a = load_a_frag(A, K, arow, kb, lane);
    v16h b = load_b_frag(Bt, K, bcol, kb, lane);
    c = __builtin_amdgcn_wmma_f32_16x16x32_f16(false, a, false, b, (short)0, c, false, false);
  }
  int n = (nt << 4) + (lane & 15);
  int mb = (mt << 4) + ((lane >> 4) << 3);
  #pragma unroll
  for (int r = 0; r < 8; ++r)
    C[(size_t)(mb + r) * Nc + n] = c[r];
}

// ---------------- attention kernels ----------------
__global__ void node_attn(const float* __restrict__ h, const float* __restrict__ aw_s,
                          const float* __restrict__ aw_d, float* __restrict__ as_n,
                          float* __restrict__ ad_n, int H, int D) {
  int i = blockIdx.x * blockDim.x + threadIdx.x;
  if (i >= NN * H) return;
  int hh = i % H;
  const float* hp = h + (size_t)i * D;
  float s1 = 0.f, s2 = 0.f;
  for (int d = 0; d < D; ++d) {
    float v = hp[d];
    s1 += v * aw_s[hh * D + d];
    s2 += v * aw_d[hh * D + d];
  }
  as_n[i] = s1; ad_n[i] = s2;
}

__global__ void edge_alpha(const int* __restrict__ ei, const float* __restrict__ as_n,
                           const float* __restrict__ ad_n, float* __restrict__ alpha,
                           unsigned* __restrict__ mEnc, int H) {
  long long i = (long long)blockIdx.x * blockDim.x + threadIdx.x;
  if (i >= (long long)EE * H) return;
  int e = (int)(i / H), hh = (int)(i % H);
  int s, d;
  if (e < NE) { s = ei[e]; d = ei[NE + e]; } else { s = d = e - NE; }
  float a = as_n[s * H + hh] + ad_n[d * H + hh];
  a = a > 0.f ? a : NEG_SLOPE * a;
  alpha[i] = a;
  atomicMax(&mEnc[d * H + hh], encf(a));
}

__global__ void edge_exp(const int* __restrict__ ei, float* __restrict__ alpha,
                         const unsigned* __restrict__ mEnc, float* __restrict__ ssum, int H) {
  long long i = (long long)blockIdx.x * blockDim.x + threadIdx.x;
  if (i >= (long long)EE * H) return;
  int e = (int)(i / H), hh = (int)(i % H);
  int d;
  if (e < NE) { d = ei[NE + e]; } else { d = e - NE; }
  float ex = expf(alpha[i] - decf(mEnc[d * H + hh]));
  alpha[i] = ex;                       // reuse buffer: now holds exp(alpha - m)
  atomicAdd(&ssum[d * H + hh], ex);
}

// block = edge, thread = feature (H*D threads)
__global__ void edge_agg(const int* __restrict__ ei, const float* __restrict__ h,
                         const float* __restrict__ ex, const float* __restrict__ ssum,
                         float* __restrict__ agg, int H, int D) {
  int e = blockIdx.x, f = threadIdx.x;
  int s, d;
  if (e < NE) { s = ei[e]; d = ei[NE + e]; } else { s = d = e - NE; }
  int hh = f / D;
  float coef = ex[(size_t)e * H + hh] / (ssum[d * H + hh] + 1e-16f);
  atomicAdd(&agg[(size_t)d * H * D + f], h[(size_t)s * H * D + f] * coef);
}

// z = elu(bn(agg + bias)), write f16
__global__ void bn_elu_f16(const float* __restrict__ agg, const float* __restrict__ bias,
                           const float* __restrict__ g, const float* __restrict__ b,
                           const float* __restrict__ mu, const float* __restrict__ var,
                           _Float16* __restrict__ zf, int total, int F) {
  int i = blockIdx.x * blockDim.x + threadIdx.x;
  if (i >= total) return;
  int f = i % F;
  float v = agg[i] + bias[f];
  v = (v - mu[f]) * rsqrtf(var[f] + BN_EPS) * g[f] + b[f];
  v = v > 0.f ? v : expm1f(v);
  zf[i] = (_Float16)v;
}

// ---------------- fused pair-MLP head ----------------
// per wave: 16 pairs; A = [z[src] | z[dst]] (16x128 f16), 4 WMMA N-tiles vs hW1t,
// ELU + dot(hW2) in regs, half-wave shuffle reduce, sigmoid, 1 float per pair out.
__global__ void __launch_bounds__(128) head_kernel(const _Float16* __restrict__ zf,
                                                   const int* __restrict__ src,
                                                   const int* __restrict__ dst,
                                                   const _Float16* __restrict__ hW1t,
                                                   const float* __restrict__ hb1,
                                                   const float* __restrict__ hW2,
                                                   const float* __restrict__ hb2,
                                                   float* __restrict__ out) {
  int wave = blockIdx.x * (blockDim.x >> 5) + (threadIdx.x >> 5);
  int lane = threadIdx.x & 31;
  if (wave >= (NP >> 4)) return;
  int row = lane & 15;
  int p = (wave << 4) + row;
  int sp = src[p], dp = dst[p];
  int kg = ((lane >> 4) << 3);
  v8f c[4] = {v8f{}, v8f{}, v8f{}, v8f{}};
  #pragma unroll
  for (int kb = 0; kb < 128; kb += 32) {
    int node = (kb < 64) ? sp : dp;
    int off = kb & 63;
    const _Float16* pa = zf + (size_t)node * OUT_DIM + off + kg;
    v8h lo = *(const v8h*)pa;
    v8h hi = *(const v8h*)(pa + 16);
    v16h a = __builtin_shufflevector(lo, hi, 0,1,2,3,4,5,6,7,8,9,10,11,12,13,14,15);
    #pragma unroll
    for (int nt = 0; nt < 4; ++nt) {
      v16h b = load_b_frag(hW1t, 128, (nt << 4) + row, kb, lane);
      c[nt] = __builtin_amdgcn_wmma_f32_16x16x32_f16(false, a, false, b, (short)0, c[nt],
                                                     false, false);
    }
  }
  float rs[8];
  #pragma unroll
  for (int r = 0; r < 8; ++r) rs[r] = 0.f;
  #pragma unroll
  for (int nt = 0; nt < 4; ++nt) {
    int n = (nt << 4) + row;
    float bb = hb1[n], w2 = hW2[n];
    #pragma unroll
    for (int r = 0; r < 8; ++r) {
      float v = c[nt][r] + bb;
      v = v > 0.f ? v : expm1f(v);     // ELU
      rs[r] += v * w2;
    }
  }
  #pragma unroll
  for (int m = 1; m < 16; m <<= 1)
    #pragma unroll
    for (int r = 0; r < 8; ++r) rs[r] += __shfl_xor(rs[r], m, 32);
  if (row == 0) {
    float hb2v = hb2[0];
    int mb = (wave << 4) + ((lane >> 4) << 3);
    #pragma unroll
    for (int r = 0; r < 8; ++r) {
      float xv = rs[r] + hb2v;
      out[mb + r] = 1.0f / (1.0f + expf(-xv));
    }
  }
}

// ---------------- host launch ----------------
#define CDIV(a, b) (((a) + (b) - 1) / (b))

extern "C" void kernel_launch(void* const* d_in, const int* in_sizes, int n_in,
                              void* d_out, int out_size, void* d_ws, size_t ws_size,
                              hipStream_t stream) {
  const float* x    = (const float*)d_in[0];
  const int*   ei   = (const int*)d_in[1];
  const int*   src  = (const int*)d_in[2];
  const int*   dst  = (const int*)d_in[3];
  const float* W1   = (const float*)d_in[4];
  const float* a1s  = (const float*)d_in[5];
  const float* a1d  = (const float*)d_in[6];
  const float* b1   = (const float*)d_in[7];
  const float* bn1g = (const float*)d_in[8];
  const float* bn1b = (const float*)d_in[9];
  const float* bn1m = (const float*)d_in[10];
  const float* bn1v = (const float*)d_in[11];
  const float* W2   = (const float*)d_in[12];
  const float* a2s  = (const float*)d_in[13];
  const float* a2d  = (const float*)d_in[14];
  const float* b2   = (const float*)d_in[15];
  const float* bn2g = (const float*)d_in[16];
  const float* bn2b = (const float*)d_in[17];
  const float* bn2m = (const float*)d_in[18];
  const float* bn2v = (const float*)d_in[19];
  const float* hW1  = (const float*)d_in[20];
  const float* hb1  = (const float*)d_in[21];
  const float* hW2  = (const float*)d_in[22];
  const float* hb2  = (const float*)d_in[23];
  float* out = (float*)d_out;
  char* ws = (char*)d_ws;

  // bump allocator with aliasing
  size_t o = 0;
  auto alloc = [&](size_t bytes) { size_t r = o; o = (o + bytes + 255) & ~(size_t)255; return r; };
  size_t off_xf16  = alloc((size_t)NN * F1 * 2);        // x f16; later reused as hpre2 (NN*64*4)
  size_t off_z1f16 = alloc((size_t)NN * F1 * 2);        // z1 f16
  size_t off_W1t   = alloc((size_t)IN_DIM * F1 * 2);
  size_t off_W2t   = alloc((size_t)F1 * OUT_DIM * 2);
  size_t off_hW1t  = alloc((size_t)(2 * OUT_DIM) * OUT_DIM * 2);
  size_t off_h1    = alloc((size_t)NN * F1 * 4);        // hpre1; later reused as agg2 (NN*64*4)
  size_t off_agg1  = alloc((size_t)NN * F1 * 4);        // agg1; later reused as z2 f16 (NN*64*2)
  size_t off_alpha = alloc((size_t)EE * HEADS * 4);     // alpha/exp conv1; reused conv2
  size_t off_as    = alloc((size_t)NN * HEADS * 4);
  size_t off_ad    = alloc((size_t)NN * HEADS * 4);
  size_t off_mEnc  = alloc((size_t)NN * HEADS * 4);
  size_t off_ssum  = alloc((size_t)NN * HEADS * 4);
  (void)ws_size; (void)in_sizes; (void)n_in; (void)out_size;

  _Float16* xf16  = (_Float16*)(ws + off_xf16);
  _Float16* z1f16 = (_Float16*)(ws + off_z1f16);
  _Float16* W1t   = (_Float16*)(ws + off_W1t);
  _Float16* W2t   = (_Float16*)(ws + off_W2t);
  _Float16* hW1t  = (_Float16*)(ws + off_hW1t);
  float*    h1    = (float*)(ws + off_h1);
  float*    agg1  = (float*)(ws + off_agg1);
  float*    alpha = (float*)(ws + off_alpha);
  float*    as_n  = (float*)(ws + off_as);
  float*    ad_n  = (float*)(ws + off_ad);
  unsigned* mEnc  = (unsigned*)(ws + off_mEnc);
  float*    ssum  = (float*)(ws + off_ssum);
  float*    h2    = (float*)(ws + off_xf16);   // alias (xf16 dead after GEMM1)
  float*    agg2  = (float*)(ws + off_h1);     // alias (hpre1 dead after agg1)
  _Float16* z2f16 = (_Float16*)(ws + off_agg1);// alias (agg1 dead after bn_elu1)

  // ---- prep: f16 conversions / weight transposes ----
  cvt_f16<<<CDIV(NN * F1, 256), 256, 0, stream>>>(x, xf16, NN * F1);
  wtrans_f16<<<CDIV(IN_DIM * F1, 256), 256, 0, stream>>>(W1, W1t, IN_DIM, F1);
  wtrans_f16<<<CDIV(F1 * OUT_DIM, 256), 256, 0, stream>>>(W2, W2t, F1, OUT_DIM);
  wtrans_f16<<<CDIV(2 * OUT_DIM * OUT_DIM, 256), 256, 0, stream>>>(hW1, hW1t, 2 * OUT_DIM, OUT_DIM);

  // ---- conv1 ----
  wmma_gemm<<<CDIV((NN / 16) * (F1 / 16), 4), 128, 0, stream>>>(xf16, W1t, h1, NN, F1, IN_DIM);
  node_attn<<<CDIV(NN * HEADS, 256), 256, 0, stream>>>(h1, a1s, a1d, as_n, ad_n, HEADS, HID);
  hipMemsetAsync(ws + off_mEnc, 0, (size_t)NN * HEADS * 4, stream);
  hipMemsetAsync(ws + off_ssum, 0, (size_t)NN * HEADS * 4, stream);
  hipMemsetAsync(ws + off_agg1, 0, (size_t)NN * F1 * 4, stream);
  {
    long long tot = (long long)EE * HEADS;
    edge_alpha<<<(unsigned)CDIV(tot, 256), 256, 0, stream>>>(ei, as_n, ad_n, alpha, mEnc, HEADS);
    edge_exp<<<(unsigned)CDIV(tot, 256), 256, 0, stream>>>(ei, alpha, mEnc, ssum, HEADS);
  }
  edge_agg<<<EE, HEADS * HID, 0, stream>>>(ei, h1, alpha, ssum, agg1, HEADS, HID);
  bn_elu_f16<<<CDIV(NN * F1, 256), 256, 0, stream>>>(agg1, b1, bn1g, bn1b, bn1m, bn1v,
                                                     z1f16, NN * F1, F1);

  // ---- conv2 (H=1, D=64) ----
  wmma_gemm<<<CDIV((NN / 16) * (OUT_DIM / 16), 4), 128, 0, stream>>>(z1f16, W2t, h2, NN, OUT_DIM, F1);
  node_attn<<<CDIV(NN, 256), 256, 0, stream>>>(h2, a2s, a2d, as_n, ad_n, 1, OUT_DIM);
  hipMemsetAsync(ws + off_mEnc, 0, (size_t)NN * 4, stream);
  hipMemsetAsync(ws + off_ssum, 0, (size_t)NN * 4, stream);
  hipMemsetAsync(ws + off_h1, 0, (size_t)NN * OUT_DIM * 4, stream);
  edge_alpha<<<(unsigned)CDIV((long long)EE, 256), 256, 0, stream>>>(ei, as_n, ad_n, alpha, mEnc, 1);
  edge_exp<<<(unsigned)CDIV((long long)EE, 256), 256, 0, stream>>>(ei, alpha, mEnc, ssum, 1);
  edge_agg<<<EE, OUT_DIM, 0, stream>>>(ei, h2, alpha, ssum, agg2, 1, OUT_DIM);
  bn_elu_f16<<<CDIV(NN * OUT_DIM, 256), 256, 0, stream>>>(agg2, b2, bn2g, bn2b, bn2m, bn2v,
                                                          z2f16, NN * OUT_DIM, OUT_DIM);

  // ---- fused pair-MLP head ----
  head_kernel<<<CDIV(NP / 16, 4), 128, 0, stream>>>(z2f16, src, dst, hW1t, hb1, hW2, hb2, out);
}
